// Attention_5093831213465
// MI455X (gfx1250) — matrix operand order repaired and checked by reading, not compiled, not run on hardware
//
#include <hip/hip_runtime.h>
#include <stdint.h>
#include <stddef.h>

// ---------------------------------------------------------------------------
// Problem constants (from reference): B=16, C=256, H=W=64
// ---------------------------------------------------------------------------
#define BB   16
#define CC   256
#define HWW  4096      // 64*64
#define CAA  32        // C/8
#define CSS  128       // C/2
#define NKEY 1024      // hw/4

typedef __attribute__((ext_vector_type(16))) __bf16 v16bf;
typedef __attribute__((ext_vector_type(8)))  float  v8f;

union FragBF {
  v16bf  v;
  __bf16 e[16];
};

__device__ __forceinline__ v8f wmma_bf16(const FragBF& a, const FragBF& b, v8f c) {
  // D(16x16 f32) = A(16x32 bf16) * B(32x16 bf16) + C
  return __builtin_amdgcn_wmma_f32_16x16x32_bf16(
      false, a.v, false, b.v, (short)0, c, false, false);
}

// A-fragment element->K mapping used by this code (B uses the same order):
//   lane<16: row=lane,  elements 0..15 hold K 0..15
//   lane>=16: row=lane-16, K 16..31
__device__ __forceinline__ int a_kidx(int e, int khalf) {
  return 2 * khalf + e;
}

// ---------------------------------------------------------------------------
// Kernel 0: one-shot weight conversion f32 -> bf16 (160 KB total).
// Removes per-workgroup reconversion from the GEMM kernels.
// ---------------------------------------------------------------------------
__global__ __launch_bounds__(256)
void attn5093_cvtw(const float* __restrict__ Wq, const float* __restrict__ Wk,
                   const float* __restrict__ Wv, const float* __restrict__ Wa,
                   __bf16* __restrict__ wqb, __bf16* __restrict__ wkb,
                   __bf16* __restrict__ wvb, __bf16* __restrict__ wab)
{
  const int i = blockIdx.x * 256 + threadIdx.x;
  if (i < 8192)        wqb[i]         = (__bf16)Wq[i];          // 32*256
  else if (i < 16384)  wkb[i - 8192]  = (__bf16)Wk[i - 8192];   // 32*256
  else if (i < 49152)  wvb[i - 16384] = (__bf16)Wv[i - 16384];  // 128*256
  else if (i < 81920)  wab[i - 49152] = (__bf16)Wa[i - 49152];  // 256*128
}

// ---------------------------------------------------------------------------
// Kernel 1: q/k/v projections. Per wave: one 16-column spatial strip,
// 12 output M-tiles (q:0..1, k:2..3, v:4..11), K=256 in 8 WMMA steps.
// x converted f32->bf16 on the fly (each element touched exactly once);
// weights already bf16.
// ---------------------------------------------------------------------------
__global__ __launch_bounds__(256)
void attn5093_proj(const float*  __restrict__ x,
                   const __bf16* __restrict__ wqb,
                   const __bf16* __restrict__ wkb,
                   const __bf16* __restrict__ wvb,
                   __bf16* __restrict__ qbf,  // [B][CA][HW]
                   __bf16* __restrict__ kf,   // [B][CA][HW]
                   __bf16* __restrict__ vf)   // [B][CS][HW]
{
  const int wg   = blockIdx.x;          // B * 32 blocks
  const int b    = wg >> 5;
  const int nb   = wg & 31;             // 128-column block
  const int wave = threadIdx.x >> 5;
  const int lane = threadIdx.x & 31;
  const int s0   = nb * 128 + wave * 16;
  const int arow   = lane & 15;
  const int bcol   = lane & 15;
  const int khalfA = (lane >> 4) * 8;
  const int kbaseB = (lane >> 4) * 16;

  const float* xb = x + (size_t)b * CC * HWW;

  v8f acc[12];
  const v8f vz = {0.f,0.f,0.f,0.f,0.f,0.f,0.f,0.f};
  #pragma unroll
  for (int i = 0; i < 12; ++i) acc[i] = vz;

  for (int ks = 0; ks < 8; ++ks) {
    const int kb = ks * 32;
    FragBF Bf;
    #pragma unroll
    for (int e = 0; e < 16; ++e) {
      const int kc = kb + kbaseB + e;                 // input channel
      Bf.e[e] = (__bf16)xb[(size_t)kc * HWW + s0 + bcol];
    }
    #pragma unroll
    for (int mt = 0; mt < 12; ++mt) {
      const __bf16* Wsel; int mrow;
      if (mt < 2)      { Wsel = wqb; mrow = mt * 16; }
      else if (mt < 4) { Wsel = wkb; mrow = (mt - 2) * 16; }
      else             { Wsel = wvb; mrow = (mt - 4) * 16; }
      FragBF Af;
      #pragma unroll
      for (int e = 0; e < 16; ++e) {
        const int kk = a_kidx(e, khalfA);
        Af.e[e] = Wsel[(size_t)(mrow + arow) * CC + kb + kk];
      }
      acc[mt] = wmma_bf16(Af, Bf, acc[mt]);
    }
  }

  const int dm = (lane >> 4) * 8;
  const int dn = lane & 15;
  #pragma unroll
  for (int mt = 0; mt < 12; ++mt) {
    __bf16* dst; int mrow;
    if (mt < 2)      { dst = qbf + (size_t)b * CAA * HWW; mrow = mt * 16; }
    else if (mt < 4) { dst = kf  + (size_t)b * CAA * HWW; mrow = (mt - 2) * 16; }
    else             { dst = vf  + (size_t)b * CSS * HWW; mrow = (mt - 4) * 16; }
    #pragma unroll
    for (int j = 0; j < 8; ++j)
      dst[(size_t)(mrow + j + dm) * HWW + s0 + dn] = (__bf16)acc[mt][j];
  }
}

// ---------------------------------------------------------------------------
// Kernel 2: 2x2 max pool. kpT stored position-major [B][1024][CA] so the
// attention A-fragments read contiguously; vp stays channel-major.
// ---------------------------------------------------------------------------
__global__ __launch_bounds__(256)
void attn5093_pool(const __bf16* __restrict__ kf,
                   const __bf16* __restrict__ vf,
                   __bf16* __restrict__ kpT,  // [B][1024][CA]
                   __bf16* __restrict__ vp)   // [B][CS][1024]
{
  const int idx = blockIdx.x * blockDim.x + threadIdx.x;
  const int total = BB * (CAA + CSS) * NKEY;
  if (idx >= total) return;
  const int p = idx & (NKEY - 1);
  const int t = idx >> 10;
  const int b = t / (CAA + CSS);
  const int c = t % (CAA + CSS);
  const int h2 = p >> 5, w2 = p & 31;
  const size_t sbase = (size_t)(2 * h2) * 64 + 2 * w2;
  if (c < CAA) {
    const __bf16* src = kf + ((size_t)b * CAA + c) * HWW + sbase;
    float m = (float)src[0];
    m = fmaxf(m, (float)src[1]);
    m = fmaxf(m, (float)src[64]);
    m = fmaxf(m, (float)src[65]);
    kpT[((size_t)b * NKEY + p) * CAA + c] = (__bf16)m;
  } else {
    const int cv = c - CAA;
    const __bf16* src = vf + ((size_t)b * CSS + cv) * HWW + sbase;
    float m = (float)src[0];
    m = fmaxf(m, (float)src[1]);
    m = fmaxf(m, (float)src[64]);
    m = fmaxf(m, (float)src[65]);
    vp[((size_t)b * CSS + cv) * NKEY + p] = (__bf16)m;
  }
}

// ---------------------------------------------------------------------------
// Kernel 3: fused attention. One WG = (batch b, 64-query block).
// LDS: S[1024][64] f32 = 256 KB (CDNA5 320KB WGP LDS makes this possible).
//  P1: S = k^T q (256 WMMA tiles / WG)
//  P2: softmax stats per query column (max, 1/sum)
//  P3: out_mid = v * softmax(S)  -- P built on the fly while loading B frags
//  P4: out = gamma * (Wa * out_mid) + x   (mid tile overlaid on S LDS)
// ---------------------------------------------------------------------------
__global__ __launch_bounds__(256)
void attn5093_attn(const __bf16* __restrict__ qbf,
                   const __bf16* __restrict__ kpT,
                   const __bf16* __restrict__ vp,
                   const __bf16* __restrict__ wab,
                   const float*  __restrict__ x,
                   const float*  __restrict__ gamma,
                   float* __restrict__ out)
{
  extern __shared__ char smem[];
  float* S  = (float*)smem;                          // [1024][64]
  float* mx = (float*)(smem + (size_t)NKEY * 64 * 4);// [64]
  float* rs = mx + 64;                               // [64]
  __shared__ float red[4][64];

  const int wg   = blockIdx.x;            // B * 64
  const int b    = wg >> 6;
  const int qblk = wg & 63;
  const int q0abs = qblk * 64;
  const int wave = threadIdx.x >> 5;
  const int lane = threadIdx.x & 31;
  const int nt   = wave & 3;              // this wave's 16-query tile
  const int qn0  = nt * 16;
  const int arow   = lane & 15;
  const int bcol   = lane & 15;
  const int khalfA = (lane >> 4) * 8;
  const int kbaseB = (lane >> 4) * 16;
  const int dm = (lane >> 4) * 8;
  const int dn = lane & 15;

  const __bf16* qb  = qbf + (size_t)b * CAA * HWW;
  const __bf16* kT  = kpT + (size_t)b * NKEY * CAA;
  const __bf16* vpb = vp  + (size_t)b * CSS * NKEY;

  // ---- Phase 1: scores S[k][q] -------------------------------------------
  const int ktile0 = (wave >> 2) * 32;
  for (int kt = 0; kt < 32; ++kt) {
    const int p0 = (ktile0 + kt) * 16;
    FragBF Af, Bf;
    #pragma unroll
    for (int e = 0; e < 16; ++e) {
      const int kk = a_kidx(e, khalfA);                    // channel 0..31
      Af.e[e] = kT[(size_t)(p0 + arow) * CAA + kk];
    }
    #pragma unroll
    for (int e = 0; e < 16; ++e) {
      const int kc = kbaseB + e;                            // channel 0..31
      Bf.e[e] = qb[(size_t)kc * HWW + q0abs + qn0 + bcol];
    }
    v8f d = {0.f,0.f,0.f,0.f,0.f,0.f,0.f,0.f};
    d = wmma_bf16(Af, Bf, d);
    #pragma unroll
    for (int j = 0; j < 8; ++j)
      S[(size_t)(p0 + j + dm) * 64 + qn0 + dn] = d[j];
  }
  __syncthreads();

  // ---- Phase 2: per-query softmax stats over 1024 keys -------------------
  {
    const int tq = threadIdx.x & 63;
    const int part = threadIdx.x >> 6;      // 4 parts x 256 keys
    float m = -3.402823466e38f;
    for (int k = part * 256; k < (part + 1) * 256; ++k)
      m = fmaxf(m, S[(size_t)k * 64 + tq]);
    red[part][tq] = m;
    __syncthreads();
    const float mm = fmaxf(fmaxf(red[0][tq], red[1][tq]),
                           fmaxf(red[2][tq], red[3][tq]));
    float s = 0.f;
    for (int k = part * 256; k < (part + 1) * 256; ++k)
      s += __expf(S[(size_t)k * 64 + tq] - mm);
    __syncthreads();
    red[part][tq] = s;
    __syncthreads();
    if (part == 0) {
      const float ss = red[0][tq] + red[1][tq] + red[2][tq] + red[3][tq];
      mx[tq] = mm;
      rs[tq] = 1.0f / ss;
    }
  }
  __syncthreads();

  // ---- Phase 3: mid[c][q] = sum_k v[c][k] * softmax(S)[k][q] -------------
  v8f acc3[4];
  {
    const v8f vz = {0.f,0.f,0.f,0.f,0.f,0.f,0.f,0.f};
    #pragma unroll
    for (int i = 0; i < 4; ++i) acc3[i] = vz;
  }
  const int c0base = (wave >> 2) * 64;      // channel half for this wave
  const float colmax = mx[qn0 + bcol];
  const float colrs  = rs[qn0 + bcol];
  for (int ks = 0; ks < 32; ++ks) {
    const int kb = ks * 32;
    FragBF Bf;
    #pragma unroll
    for (int e = 0; e < 16; ++e) {
      const int kk = kb + kbaseB + e;
      const float sv = S[(size_t)kk * 64 + qn0 + bcol];
      Bf.e[e] = (__bf16)(__expf(sv - colmax) * colrs);  // P built on the fly
    }
    #pragma unroll
    for (int mt = 0; mt < 4; ++mt) {
      const int c0 = c0base + mt * 16;
      FragBF Af;
      #pragma unroll
      for (int e = 0; e < 16; ++e) {
        const int kk = a_kidx(e, khalfA);
        Af.e[e] = vpb[(size_t)(c0 + arow) * NKEY + kb + kk];
      }
      acc3[mt] = wmma_bf16(Af, Bf, acc3[mt]);
    }
  }
  __syncthreads();                          // all S reads done

  // mid [128][64] overlays the S buffer
  float* mid = S;
  #pragma unroll
  for (int mt = 0; mt < 4; ++mt) {
    const int c0 = c0base + mt * 16;
    #pragma unroll
    for (int j = 0; j < 8; ++j)
      mid[(size_t)(c0 + j + dm) * 64 + qn0 + dn] = acc3[mt][j];
  }
  __syncthreads();

  // ---- Phase 4: out = gamma * (Wa @ mid) + x -----------------------------
  v8f acc4[8];
  {
    const v8f vz = {0.f,0.f,0.f,0.f,0.f,0.f,0.f,0.f};
    #pragma unroll
    for (int i = 0; i < 8; ++i) acc4[i] = vz;
  }
  const int mbase4 = (wave >> 2) * 128;     // output-channel half
  for (int ks = 0; ks < 4; ++ks) {
    const int kb = ks * 32;
    FragBF Bf;
    #pragma unroll
    for (int e = 0; e < 16; ++e) {
      const int kk = kb + kbaseB + e;
      Bf.e[e] = (__bf16)mid[(size_t)kk * 64 + qn0 + bcol];
    }
    #pragma unroll
    for (int mt = 0; mt < 8; ++mt) {
      const int o0 = mbase4 + mt * 16;
      FragBF Af;
      #pragma unroll
      for (int e = 0; e < 16; ++e) {
        const int kk = a_kidx(e, khalfA);
        Af.e[e] = wab[(size_t)(o0 + arow) * CSS + kb + kk];
      }
      acc4[mt] = wmma_bf16(Af, Bf, acc4[mt]);
    }
  }
  const float g = gamma[0];
  const float* xb = x   + (size_t)b * CC * HWW;
  float*       ob = out + (size_t)b * CC * HWW;
  #pragma unroll
  for (int mt = 0; mt < 8; ++mt) {
    const int o0 = mbase4 + mt * 16;
    #pragma unroll
    for (int j = 0; j < 8; ++j) {
      const size_t oi = (size_t)(o0 + j + dm) * HWW + q0abs + qn0 + dn;
      ob[oi] = g * acc4[mt][j] + xb[oi];
    }
  }
}

// ---------------------------------------------------------------------------
// Host launcher
// ---------------------------------------------------------------------------
extern "C" void kernel_launch(void* const* d_in, const int* in_sizes, int n_in,
                              void* d_out, int out_size, void* d_ws, size_t ws_size,
                              hipStream_t stream) {
  const float* x     = (const float*)d_in[0];
  const float* Wq    = (const float*)d_in[1];
  const float* Wk    = (const float*)d_in[2];
  const float* Wv    = (const float*)d_in[3];
  const float* Wa    = (const float*)d_in[4];
  const float* gamma = (const float*)d_in[5];
  float* out = (float*)d_out;

  // bf16 staging buffers carved from workspace (~29.2 MB total)
  char* ws = (char*)d_ws;
  __bf16* qbf = (__bf16*)(ws);                                   // 4 MB
  __bf16* kf  = (__bf16*)(ws + (size_t)4  * 1024 * 1024);        // 4 MB
  __bf16* vf  = (__bf16*)(ws + (size_t)8  * 1024 * 1024);        // 16 MB
  __bf16* kpT = (__bf16*)(ws + (size_t)24 * 1024 * 1024);        // 1 MB
  __bf16* vp  = (__bf16*)(ws + (size_t)25 * 1024 * 1024);        // 4 MB
  __bf16* wqb = (__bf16*)(ws + (size_t)29 * 1024 * 1024);        // 16 KB
  __bf16* wkb = wqb + 8192;                                      // 16 KB
  __bf16* wvb = wkb + 8192;                                      // 64 KB
  __bf16* wab = wvb + 32768;                                     // 64 KB

  // Kernel 0: weight conversion (81920 elements)
  attn5093_cvtw<<<320, 256, 0, stream>>>(Wq, Wk, Wv, Wa, wqb, wkb, wvb, wab);

  // Kernel 1: projections (B * 32 column-blocks)
  attn5093_proj<<<BB * 32, 256, 0, stream>>>(x, wqb, wkb, wvb, qbf, kf, vf);

  // Kernel 2: 2x2 max pool
  {
    const int total = BB * (CAA + CSS) * NKEY;
    attn5093_pool<<<(total + 255) / 256, 256, 0, stream>>>(kf, vf, kpT, vp);
  }

  // Kernel 3: fused attention; 256 KB + 512 B dynamic LDS per WG
  const int smem_bytes = NKEY * 64 * 4 + 2 * 64 * 4;
  (void)hipFuncSetAttribute((const void*)attn5093_attn,
                            hipFuncAttributeMaxDynamicSharedMemorySize,
                            smem_bytes);
  attn5093_attn<<<BB * 64, 256, smem_bytes, stream>>>(qbf, kpT, vp, wab, x, gamma, out);
}
